// EchoNeedleRIN_75892072121023
// MI455X (gfx1250) — matrix operand order repaired and checked by reading, not compiled, not run on hardware
//
#include <hip/hip_runtime.h>

// ---------------------------------------------------------------------------
// EchoNeedle persistent-kernel implementation for gfx1250 (MI455X).
// bf16 WMMA (v_wmma_f32_16x16x32_bf16), f32 accumulation, reservoir state
// held in accumulator VGPRs across all 1024 time steps, device-scope
// grid barrier between the three dependent phases of each step.
// Imag weights pre-negated (no per-chunk VALU xors); LUT sin/cos via
// hardware V_SIN_F32/V_COS_F32 on normalized (turns) input.
// ---------------------------------------------------------------------------

typedef __bf16 bf16;
typedef bf16         v16bf __attribute__((ext_vector_type(16)));
typedef float        v8f   __attribute__((ext_vector_type(8)));
typedef unsigned int u32;
typedef u32          v4u   __attribute__((ext_vector_type(4)));

#define NWG    128
#define NTHR   256
#define SEQ    1024
#define BATCH  128
#define DMODEL 256
#define NRES   1024

constexpr float PHI_F     = 1.618033988749895f;
constexpr float TWO_PI_F  = 6.2831853071795864769f;
constexpr float INV_2PI_F = 0.15915494309189533577f;

// ---- workspace layout (all offsets 256B aligned) --------------------------
constexpr size_t SZ_X    = (size_t)BATCH * DMODEL * sizeof(bf16);   // 64 KB
constexpr size_t SZ_N    = (size_t)BATCH * NRES   * sizeof(bf16);   // 256 KB
constexpr size_t SZ_WT   = (size_t)2 * NRES * DMODEL * sizeof(bf16);// 1 MB (both layers)
constexpr size_t SZ_PQ   = (size_t)DMODEL * 256 * sizeof(bf16);     // 128 KB

constexpr size_t OFF_BAR    = 0;
constexpr size_t OFF_X0R    = 256;
constexpr size_t OFF_X0I    = OFF_X0R    + SZ_X;
constexpr size_t OFF_X1R    = OFF_X0I    + SZ_X;
constexpr size_t OFF_X1I    = OFF_X1R    + SZ_X;
constexpr size_t OFF_X2R    = OFF_X1I    + SZ_X;
constexpr size_t OFF_X2I    = OFF_X2R    + SZ_X;
constexpr size_t OFF_N0R    = OFF_X2I    + SZ_X;
constexpr size_t OFF_N0I    = OFF_N0R    + SZ_N;
constexpr size_t OFF_N1R    = OFF_N0I    + SZ_N;
constexpr size_t OFF_N1I    = OFF_N1R    + SZ_N;
constexpr size_t OFF_WINTR  = OFF_N1I    + SZ_N;   // winT  [l][n][k] bf16
constexpr size_t OFF_WINTI  = OFF_WINTR  + SZ_WT;
constexpr size_t OFF_WINTIN = OFF_WINTI  + SZ_WT;  // -win_i (pre-negated)
constexpr size_t OFF_WOUTTR = OFF_WINTIN + SZ_WT;  // woutT [l][d][k] bf16
constexpr size_t OFF_WOUTTI = OFF_WOUTTR + SZ_WT;
constexpr size_t OFF_WOUTTIN= OFF_WOUTTI + SZ_WT;  // -wout_i (pre-negated)
constexpr size_t OFF_P      = OFF_WOUTTIN+ SZ_WT;  // (outp_r+outp_i)^T [s][k]
constexpr size_t OFF_Q      = OFF_P      + SZ_PQ;  // (outp_r-outp_i)^T [s][k]

// ---------------------------------------------------------------------------
// LUT sin/cos: the reference quantizes the wrapped phase onto a 4096-entry
// grid of one turn. V_SIN_F32 / V_COS_F32 take normalized (turns) input:
// sin(2*pi*x) -- so the quantized fraction feeds the trans unit directly.
__device__ static inline void lut_sincos(float th, float& s, float& c) {
    float t = th * INV_2PI_F;
    t -= floorf(t);                                   // frac -> [0,1) turns
    float q = floorf(t * 4096.0f) * (1.0f / 4096.0f); // LUT bin, in turns
    s = __builtin_amdgcn_sinf(q);
    c = __builtin_amdgcn_cosf(q);
}

__device__ static inline v8f wmma_bf16(v16bf a, v16bf b, v8f c) {
    return __builtin_amdgcn_wmma_f32_16x16x32_bf16(
        false, a, false, b, (short)0, c, false, false);
}

// A fragment (16x32 bf16), row-major src [rows][ld]:
// lane m=L&15, khalf=L>>4: VGPR0..3 = K[kb+8*kh .. +7], VGPR4..7 = K[kb+16+8*kh .. +7]
__device__ static inline v16bf load_a_frag(const bf16* __restrict__ src, int ld,
                                           int row0, int kb, int lane) {
    const bf16* p = src + (size_t)(row0 + (lane & 15)) * ld + kb + ((lane >> 4) << 3);
    union { v4u u[2]; v16bf v; } x;
    x.u[0] = *(const v4u*)(p);
    x.u[1] = *(const v4u*)(p + 16);
    return x.v;
}

// B fragment (32x16 bf16) from pre-transposed weights bt[n][k]:
// lane n=L&15, khalf=L>>4: VGPR0..3 = K[kb+8*kh .. +7], VGPR4..7 = K[kb+16+8*kh .. +7]
__device__ static inline v16bf load_b_frag(const bf16* __restrict__ bt, int ldk,
                                           int col0, int kb, int lane) {
    const bf16* p = bt + (size_t)(col0 + (lane & 15)) * ldk + kb + ((lane >> 4) << 3);
    union { v4u u[2]; v16bf v; } x;
    x.u[0] = *(const v4u*)(p);
    x.u[1] = *(const v4u*)(p + 16);
    return x.v;
}

// ---- device-scope sense-reversing grid barrier ----------------------------
__device__ static inline void grid_sync(unsigned* bar) {
    __threadfence();
    __syncthreads();
    if (threadIdx.x == 0) {
        unsigned gen  = __hip_atomic_load(bar + 1, __ATOMIC_ACQUIRE, __HIP_MEMORY_SCOPE_AGENT);
        unsigned prev = __hip_atomic_fetch_add(bar, 1u, __ATOMIC_ACQ_REL, __HIP_MEMORY_SCOPE_AGENT);
        if (prev == (unsigned)(NWG - 1)) {
            __hip_atomic_store(bar, 0u, __ATOMIC_RELAXED, __HIP_MEMORY_SCOPE_AGENT);
            __hip_atomic_fetch_add(bar + 1, 1u, __ATOMIC_RELEASE, __HIP_MEMORY_SCOPE_AGENT);
        } else {
            while (__hip_atomic_load(bar + 1, __ATOMIC_ACQUIRE, __HIP_MEMORY_SCOPE_AGENT) == gen)
                __builtin_amdgcn_s_sleep(2);
        }
    }
    __syncthreads();
}

// ---- elementwise h recurrence (euler_transform), one element per call -----
__device__ static inline void stepA(float& hr, float& hi, int e, int t,
                                    const int* __restrict__ ids,
                                    const float* __restrict__ emb,
                                    bf16* __restrict__ x0r, bf16* __restrict__ x0i) {
    const int b = e >> 8, d = e & 255;
    const int tok = ids[b * SEQ + t];
    const float w  = emb[(size_t)tok * 512 + d];
    const float bb = emb[(size_t)tok * 512 + 256 + d];
    const float inv_wl = 1.0f / (1.0f + fabsf(w));
    // wrap(t*PHI) == (frac(t*PHI/2pi))*2pi  (jnp.mod for non-negative t)
    const float u = (float)t * (PHI_F * INV_2PI_F);
    const float tphi = (u - floorf(u)) * TWO_PI_F;
    float sr, cr, si, ci;
    lut_sincos(hr * inv_wl + bb + tphi, sr, cr);
    lut_sincos(hi * inv_wl + bb + tphi, si, ci);
    hr = cr * ci - sr * si;
    hi = cr * si + sr * ci;
    x0r[e] = (bf16)hr;   // e == b*DMODEL + d
    x0i[e] = (bf16)hi;
}

// ---- up-projection GEMM + reservoir rotation; state lives in sr/si -------
// btiN is the pre-negated imaginary weight pack: u_r = xr*Wr + xi*(-Wi)
__device__ static inline void layer_u(const bf16* __restrict__ xr,
                                      const bf16* __restrict__ xi,
                                      const bf16* __restrict__ btr,
                                      const bf16* __restrict__ bti,
                                      const bf16* __restrict__ btiN,
                                      int bt, int nt, int lane,
                                      float om, float tf,
                                      v8f& sr, v8f& si,
                                      bf16* __restrict__ nrow_r,
                                      bf16* __restrict__ nrow_i, bool store_n) {
    v8f ur = {0.f,0.f,0.f,0.f,0.f,0.f,0.f,0.f};
    v8f ui = {0.f,0.f,0.f,0.f,0.f,0.f,0.f,0.f};
#pragma unroll
    for (int kc = 0; kc < 8; ++kc) {                 // K = 256
        const int kb = kc * 32;
        v16bf a_r  = load_a_frag(xr, DMODEL, bt * 16, kb, lane);
        v16bf a_i  = load_a_frag(xi, DMODEL, bt * 16, kb, lane);
        v16bf b_r  = load_b_frag(btr,  DMODEL, nt * 16, kb, lane);
        v16bf b_i  = load_b_frag(bti,  DMODEL, nt * 16, kb, lane);
        v16bf b_in = load_b_frag(btiN, DMODEL, nt * 16, kb, lane);
        ur = wmma_bf16(a_r, b_r, ur);
        ur = wmma_bf16(a_i, b_in, ur);               // u_r = xr*Wr - xi*Wi
        ui = wmma_bf16(a_r, b_i, ui);
        ui = wmma_bf16(a_i, b_r, ui);                // u_i = xr*Wi + xi*Wr
    }
    float sn, cn;
    lut_sincos(om + tf * PHI_F, sn, cn);
#pragma unroll
    for (int v = 0; v < 8; ++v) {
        float nr2 = cn * sr[v] - sn * si[v] + ur[v];
        float ni2 = sn * sr[v] + cn * si[v] + ui[v];
        sr[v] = nr2; si[v] = ni2;
    }
    if (store_n) {
        const int n  = nt * 16 + (lane & 15);
        const int rb = bt * 16 + ((lane >> 4) << 3);
#pragma unroll
        for (int v = 0; v < 8; ++v) {
            nrow_r[(size_t)(rb + v) * NRES + n] = (bf16)sr[v];
            nrow_i[(size_t)(rb + v) * NRES + n] = (bf16)si[v];
        }
    }
}

// ---- down-projection: one [16x16] tile per WG, K=1024 split over 8 waves --
__device__ static inline void proj_down(const bf16* __restrict__ nr,
                                        const bf16* __restrict__ ni,
                                        const bf16* __restrict__ wtr,
                                        const bf16* __restrict__ wti,
                                        const bf16* __restrict__ wtiN,
                                        bf16* __restrict__ xr_out,
                                        bf16* __restrict__ xi_out,
                                        int wg, int wv, int lane, float* lds) {
    const int bt = wg >> 4, dt = wg & 15;
    v8f accr = {0.f,0.f,0.f,0.f,0.f,0.f,0.f,0.f};
    v8f acci = {0.f,0.f,0.f,0.f,0.f,0.f,0.f,0.f};
#pragma unroll
    for (int kc = 0; kc < 4; ++kc) {                 // 128 K per wave
        const int kb = wv * 128 + kc * 32;
        v16bf a_r  = load_a_frag(nr, NRES, bt * 16, kb, lane);
        v16bf a_i  = load_a_frag(ni, NRES, bt * 16, kb, lane);
        v16bf b_r  = load_b_frag(wtr,  NRES, dt * 16, kb, lane);
        v16bf b_i  = load_b_frag(wti,  NRES, dt * 16, kb, lane);
        v16bf b_in = load_b_frag(wtiN, NRES, dt * 16, kb, lane);
        accr = wmma_bf16(a_r, b_r, accr);
        accr = wmma_bf16(a_i, b_in, accr);
        acci = wmma_bf16(a_r, b_i, acci);
        acci = wmma_bf16(a_i, b_r, acci);
    }
    float* slot = lds + (size_t)(wv * 32 + lane) * 16;
#pragma unroll
    for (int v = 0; v < 8; ++v) { slot[v] = accr[v]; slot[8 + v] = acci[v]; }
    __syncthreads();
    if (wv == 0) {
        const int d  = dt * 16 + (lane & 15);
        const int rb = bt * 16 + ((lane >> 4) << 3);
#pragma unroll
        for (int v = 0; v < 8; ++v) {
            float ar2 = 0.f, ai2 = 0.f;
            for (int w2 = 0; w2 < 8; ++w2) {
                ar2 += lds[(size_t)(w2 * 32 + lane) * 16 + v];
                ai2 += lds[(size_t)(w2 * 32 + lane) * 16 + 8 + v];
            }
            xr_out[(size_t)(rb + v) * DMODEL + d] = (bf16)ar2;
            xi_out[(size_t)(rb + v) * DMODEL + d] = (bf16)ai2;
        }
    }
    __syncthreads();
}

// ---------------------------------------------------------------------------
__global__ __launch_bounds__(NTHR, 1)
void echo_kernel(const int*   __restrict__ ids,
                 const float* __restrict__ emb,
                 const float* __restrict__ win_r,  const float* __restrict__ win_i,
                 const float* __restrict__ wout_r, const float* __restrict__ wout_i,
                 const float* __restrict__ omega,
                 const float* __restrict__ outp_r, const float* __restrict__ outp_i,
                 float* __restrict__ out, char* __restrict__ ws) {
    __shared__ float lds_red[NTHR * 16];             // 16 KB reduction scratch

    unsigned* bar = (unsigned*)(ws + OFF_BAR);
    bf16* x0r = (bf16*)(ws + OFF_X0R);  bf16* x0i = (bf16*)(ws + OFF_X0I);
    bf16* x1r = (bf16*)(ws + OFF_X1R);  bf16* x1i = (bf16*)(ws + OFF_X1I);
    bf16* x2r = (bf16*)(ws + OFF_X2R);  bf16* x2i = (bf16*)(ws + OFF_X2I);
    bf16* n0r = (bf16*)(ws + OFF_N0R);  bf16* n0i = (bf16*)(ws + OFF_N0I);
    bf16* n1r = (bf16*)(ws + OFF_N1R);  bf16* n1i = (bf16*)(ws + OFF_N1I);
    bf16* winTr   = (bf16*)(ws + OFF_WINTR);
    bf16* winTi   = (bf16*)(ws + OFF_WINTI);
    bf16* winTiN  = (bf16*)(ws + OFF_WINTIN);
    bf16* woutTr  = (bf16*)(ws + OFF_WOUTTR);
    bf16* woutTi  = (bf16*)(ws + OFF_WOUTTI);
    bf16* woutTiN = (bf16*)(ws + OFF_WOUTTIN);
    bf16* Pp = (bf16*)(ws + OFF_P);     bf16* Qp = (bf16*)(ws + OFF_Q);

    const int wg     = blockIdx.x;        // 0..127
    const int tl     = threadIdx.x;       // 0..255
    const int wv     = tl >> 5;           // wave in WG, 0..7
    const int lane   = tl & 31;
    const int wave_g = wg * 8 + wv;       // 0..1023
    const int tid    = wg * NTHR + tl;    // 0..32767
    const int P      = NWG * NTHR;

    // ---------------- prologue: pack weights to bf16 fragment layouts ------
    for (int x = tid; x < 2 * NRES * DMODEL; x += P) {   // winT[l][n][k]
        int l = x >> 18, rem = x & 262143;
        int n = rem >> 8, k = rem & 255;
        size_t si2 = (size_t)l * 262144 + (size_t)k * NRES + n;  // win[l][k][n]
        float vi = win_i[si2];
        winTr[x]  = (bf16)win_r[si2];
        winTi[x]  = (bf16)vi;
        winTiN[x] = (bf16)(-vi);
    }
    for (int x = tid; x < 2 * NRES * DMODEL; x += P) {   // woutT[l][d][k]
        int l = x >> 18, rem = x & 262143;
        int d = rem >> 10, k = rem & 1023;
        size_t si2 = (size_t)l * 262144 + (size_t)k * DMODEL + d; // wout[l][k][d]
        float vi = wout_i[si2];
        woutTr[x]  = (bf16)wout_r[si2];
        woutTi[x]  = (bf16)vi;
        woutTiN[x] = (bf16)(-vi);
    }
    for (int x = tid; x < 256 * DMODEL; x += P) {        // P/Q [s][k]
        int s = x >> 8, k = x & 255;
        float r = outp_r[(size_t)k * 256 + s], i2 = outp_i[(size_t)k * 256 + s];
        Pp[x] = (bf16)(r + i2);
        Qp[x] = (bf16)(r - i2);
    }

    // ---------------- per-wave persistent state ----------------------------
    float hr0 = 0.f, hi0 = 0.f, hr1 = 0.f, hi1 = 0.f;   // h recurrence (wg>=64)
    int e0 = 0, e1 = 0;
    v8f s0r = {0.f,0.f,0.f,0.f,0.f,0.f,0.f,0.f}, s0i = s0r; // reservoir (wg<64)
    v8f s1r = s0r, s1i = s0r;
    float om0 = 0.f, om1 = 0.f;
    int bt = 0, nt = 0;

    if (wg < 64) {
        bt  = wave_g >> 6;                 // 0..7   batch tile
        nt  = wave_g & 63;                 // 0..63  neuron tile
        om0 = omega[nt * 16 + (lane & 15)];
        om1 = omega[NRES + nt * 16 + (lane & 15)];
    } else {
        const int aw = wave_g - 512;       // 0..511
        e0 = aw * 32 + lane;               // each lane owns 2 of 32768 elements
        e1 = e0 + 16384;
        stepA(hr0, hi0, e0, 0, ids, emb, x0r, x0i);  // x0 for t = 0
        stepA(hr1, hi1, e1, 0, ids, emb, x0r, x0i);
    }

    grid_sync(bar);

    // ---------------- the scan ---------------------------------------------
#pragma unroll 1
    for (int t = 0; t < SEQ; ++t) {
        const float tf = (float)t;

        // phase 1: u0 = x0 @ win0 (complex), rotate s0, emit n0 (bf16)
        if (wg < 64)
            layer_u(x0r, x0i, winTr, winTi, winTiN, bt, nt, lane, om0, tf,
                    s0r, s0i, n0r, n0i, true);
        grid_sync(bar);

        // phase 2: x1 = n0 @ wout0 (complex), K split across 8 waves/WG
        proj_down(n0r, n0i, woutTr, woutTi, woutTiN, x1r, x1i,
                  wg, wv, lane, lds_red);
        grid_sync(bar);

        // phase 3: u1/s1 update; concurrently compute next step's x0 (h rec.)
        if (wg < 64) {
            layer_u(x1r, x1i, winTr + 262144, winTi + 262144, winTiN + 262144,
                    bt, nt, lane, om1, tf, s1r, s1i, n1r, n1i, t == SEQ - 1);
        } else if (t + 1 < SEQ) {
            stepA(hr0, hi0, e0, t + 1, ids, emb, x0r, x0i);
            stepA(hr1, hi1, e1, t + 1, ids, emb, x0r, x0i);
        }
        grid_sync(bar);
    }

    // ---------------- epilogue: x2 = n1 @ wout1; logits = x2r*P + x2i*Q ----
    proj_down(n1r, n1i, woutTr + 262144, woutTi + 262144, woutTiN + 262144,
              x2r, x2i, wg, wv, lane, lds_red);
    grid_sync(bar);

    {
        const int bt2 = wg >> 4, st = wg & 15;    // [128 x 256] output tiles
        const int kb = wv * 32;                    // K=256 split over 8 waves
        v16bf a_r = load_a_frag(x2r, DMODEL, bt2 * 16, kb, lane);
        v16bf a_i = load_a_frag(x2i, DMODEL, bt2 * 16, kb, lane);
        v16bf b_p = load_b_frag(Pp, DMODEL, st * 16, kb, lane);
        v16bf b_q = load_b_frag(Qp, DMODEL, st * 16, kb, lane);
        v8f acc = {0.f,0.f,0.f,0.f,0.f,0.f,0.f,0.f};
        acc = wmma_bf16(a_r, b_p, acc);
        acc = wmma_bf16(a_i, b_q, acc);
        float* slot = lds_red + (size_t)tl * 16;
#pragma unroll
        for (int v = 0; v < 8; ++v) { slot[v] = acc[v]; slot[8 + v] = 0.f; }
        __syncthreads();
        if (wv == 0) {
            const int s  = st * 16 + (lane & 15);
            const int rb = bt2 * 16 + ((lane >> 4) << 3);
#pragma unroll
            for (int v = 0; v < 8; ++v) {
                float a = 0.f;
                for (int w2 = 0; w2 < 8; ++w2)
                    a += lds_red[(size_t)(w2 * 32 + lane) * 16 + v];
                out[(size_t)(rb + v) * 256 + s] = a;
            }
        }
    }
}

// ---------------------------------------------------------------------------
extern "C" void kernel_launch(void* const* d_in, const int* in_sizes, int n_in,
                              void* d_out, int out_size, void* d_ws, size_t ws_size,
                              hipStream_t stream) {
    (void)in_sizes; (void)n_in; (void)out_size; (void)ws_size;
    // reset grid barrier state (graph-capture safe)
    hipMemsetAsync(d_ws, 0, 256, stream);
    echo_kernel<<<NWG, NTHR, 0, stream>>>(
        (const int*)d_in[0],   (const float*)d_in[1],
        (const float*)d_in[2], (const float*)d_in[3],
        (const float*)d_in[4], (const float*)d_in[5],
        (const float*)d_in[6], (const float*)d_in[7], (const float*)d_in[8],
        (float*)d_out, (char*)d_ws);
}
